// PatchwiseHadamard_59785944760410
// MI455X (gfx1250) — matrix operand order repaired and checked by reading, not compiled, not run on hardware
//
#include <hip/hip_runtime.h>

// ---------------------------------------------------------------------------
// Patchwise 256x256 Hadamard sandwich on MI455X (gfx1250, wave32, WMMA).
//   Pass A:  Y   = s * (H @ X @ H) + t      (x     -> d_out)
//   Pass B:  OUT = H @ Y @ H                (d_out -> d_out, per-patch in-place)
//
// H entries +-1/16 are EXACT bf16; data uses a hi+lo bf16 split (2 WMMAs per
// K-chunk, ~16-17 bit accuracy).  Splits happen ONCE at load/stage time and
// are stored in LDS pre-packed in the WMMA operand pair layout, column-major
// along K, so every operand fetch in the hot loop is 2x ds_load_b128 feeding
// v_wmma directly (zero VALU).  H operand tiles are synthesized from per-lane
// Walsh words + scalar block parities (Sylvester recursion).
// ---------------------------------------------------------------------------

typedef __attribute__((ext_vector_type(16))) __bf16 v16bf;
typedef __attribute__((ext_vector_type(8)))  float  v8f;

#define IMGW 2048
#define XW   132     // uint stride per X column (128 pair-rows + pad): b128-clean
#define TW   132     // uint stride per T row    (128 pair-cols + pad): b128-clean

union BV { v16bf v; unsigned u[8]; uint4 q[2]; };

// split f32 -> (hi bf16 bits, lo bf16 bits), round-to-nearest-even
__device__ __forceinline__ void split1(float x, unsigned& hb, unsigned& lb) {
    union { float f; unsigned u; } v; v.f = x;
    unsigned hf = (v.u + 0x7FFFu + ((v.u >> 16) & 1u)) & 0xFFFF0000u; // hi as f32 bits
    union { unsigned u; float f; } hv; hv.u = hf;
    union { float f; unsigned u; } l; l.f = x - hv.f;
    hb = hf >> 16;
    lb = (l.u + 0x7FFFu + ((l.u >> 16) & 1u)) >> 16;
}

#define WMMA(ACC, A, B) \
    ACC = __builtin_amdgcn_wmma_f32_16x16x32_bf16(false, (A).v, false, (B).v, (short)0, ACC, false, false)

template<bool AFFINE>
__global__ __launch_bounds__(256, 1)
void hadamard_pass(const float* __restrict__ src, float* __restrict__ dst,
                   const float* __restrict__ cw, const float* __restrict__ cb,
                   const float* __restrict__ ow, const float* __restrict__ ob)
{
    // X packed column-major: Xhi/Xlo[c*XW + m] = bf16(X[2m][c]) | bf16(X[2m+1][c])<<16
    __shared__ unsigned Xhi[256 * XW];   // 135168 B
    __shared__ unsigned Xlo[256 * XW];   // 135168 B
    // T packed row-major, pairs along columns (GEMM2's K): Thi[row*TW + k/2]
    __shared__ unsigned Thi[32 * TW];    //  16896 B
    __shared__ unsigned Tlo[32 * TW];    //  16896 B   total 304128 B <= 320 KB

    const int tid  = threadIdx.x;
    const int wave = tid >> 5;
    const int lane = tid & 31;
    const int half = lane >> 4;
    const int l15  = lane & 15;

    const int blk = blockIdx.x;
    const int img = blk >> 6, hp = (blk >> 3) & 7, wp = blk & 7;
    const size_t base = (size_t)img * IMGW * IMGW + (size_t)hp * 256 * IMGW + (size_t)wp * 256;

    float s = 1.f, t = 0.f;
    if (AFFINE) {
        s = 0.f;
        #pragma unroll
        for (int c = 0; c < 16; ++c) { s += ow[c] * cw[c]; t += ow[c] * cb[c]; }
        t += ob[0];
    }

    // per-lane Walsh word: bit b = parity(popc(b & l15)), b = 0..15
    unsigned w16 = 0;
    #pragma unroll
    for (int b = 0; b < 16; ++b) w16 |= (unsigned)(__builtin_popcount(b & l15) & 1) << b;
    const unsigned W32 = w16 | (w16 << 16);
    const unsigned HBF = 0x3D803D80u;                 // packed +1/16 bf16 pair

    // ---- load patch (coalesced float4 rows), split ONCE, store column-major pairs ----
    for (int it = 0; it < 16; ++it) {
        int idx = it * 256 + tid;
        int g  = idx >> 6;            // 4-row band: rows 4g..4g+3
        int c4 = (idx & 63) << 2;     // 4 columns
        const float* p0 = src + base + (size_t)(4 * g) * IMGW + c4;
        const float4 r0 = *reinterpret_cast<const float4*>(p0);
        const float4 r1 = *reinterpret_cast<const float4*>(p0 + IMGW);
        const float4 r2 = *reinterpret_cast<const float4*>(p0 + 2 * IMGW);
        const float4 r3 = *reinterpret_cast<const float4*>(p0 + 3 * IMGW);
        const float a0[4] = {r0.x, r0.y, r0.z, r0.w};
        const float a1[4] = {r1.x, r1.y, r1.z, r1.w};
        const float a2[4] = {r2.x, r2.y, r2.z, r2.w};
        const float a3[4] = {r3.x, r3.y, r3.z, r3.w};
        #pragma unroll
        for (int j = 0; j < 4; ++j) {
            unsigned h0, l0, h1, l1, h2, l2, h3, l3;
            split1(a0[j], h0, l0); split1(a1[j], h1, l1);
            split1(a2[j], h2, l2); split1(a3[j], h3, l3);
            const int c = c4 + j;
            *reinterpret_cast<uint2*>(&Xhi[c * XW + 2 * g]) = make_uint2(h0 | (h1 << 16), h2 | (h3 << 16));
            *reinterpret_cast<uint2*>(&Xlo[c * XW + 2 * g]) = make_uint2(l0 | (l1 << 16), l2 | (l3 << 16));
        }
    }
    __syncthreads();

    for (int strip = 0; strip < 8; ++strip) {
        // ============ GEMM1: T[32x256] = H[strip rows] @ X ============
        // wave owns M-tiles mt=0,1 and N-tiles nt = 2*wave, 2*wave+1
        v8f acc[2][2] = {};
        for (int kc = 0; kc < 8; ++kc) {
            // A = H tile: sign(jl) = W32(jl) ^ parity(kc&strip) ^ (mt & jl>=16)
            const unsigned P1 = (unsigned)(__builtin_popcount(kc & strip) & 1);
            const unsigned SW = (W32 ^ (0u - P1)) >> (8 * half);
            BV a0, a1;
            #pragma unroll
            for (int v = 0; v < 8; ++v) {
                const int k0 = (v < 4) ? 2 * v : 8 + 2 * v;
                unsigned b0 = (SW >> k0) & 1u, b1 = (SW >> (k0 + 1)) & 1u;
                a0.u[v] = HBF | (b0 << 15) | (b1 << 31);
                a1.u[v] = a0.u[v] ^ ((v >= 4) ? 0x80008000u : 0u);
            }
            #pragma unroll
            for (int q = 0; q < 2; ++q) {
                const int c = (wave * 2 + q) * 16 + l15;
                const int o = c * XW + kc * 16 + half * 4;
                BV bh, bl;
                bh.q[0] = *reinterpret_cast<const uint4*>(&Xhi[o]);
                bh.q[1] = *reinterpret_cast<const uint4*>(&Xhi[o + 8]);
                bl.q[0] = *reinterpret_cast<const uint4*>(&Xlo[o]);
                bl.q[1] = *reinterpret_cast<const uint4*>(&Xlo[o + 8]);
                WMMA(acc[0][q], a0, bh); WMMA(acc[0][q], a0, bl);
                WMMA(acc[1][q], a1, bh); WMMA(acc[1][q], a1, bl);
            }
        }
        // stage T: split ONCE, store bf16 col-pairs (GEMM2 A layout)
        {
            unsigned short* th = reinterpret_cast<unsigned short*>(Thi);
            unsigned short* tl = reinterpret_cast<unsigned short*>(Tlo);
            #pragma unroll
            for (int mt = 0; mt < 2; ++mt)
                #pragma unroll
                for (int q = 0; q < 2; ++q) {
                    const int col = (wave * 2 + q) * 16 + l15;
                    #pragma unroll
                    for (int r = 0; r < 8; ++r) {
                        const int M = mt * 16 + r + 8 * half;
                        unsigned hb, lb;
                        split1(acc[mt][q][r], hb, lb);
                        th[M * 2 * TW + col] = (unsigned short)hb;
                        tl[M * 2 * TW + col] = (unsigned short)lb;
                    }
                }
        }
        __syncthreads();

        // ============ GEMM2: O[32x256] = T @ H ============
        v8f o[2][2] = {};
        for (int kc = 0; kc < 8; ++kc) {
            // A = T tiles (packed pairs): 2x ds_load_b128 per vector, zero VALU
            const int ofs0 = l15 * TW + kc * 16 + 8 * half;
            const int ofs1 = (16 + l15) * TW + kc * 16 + 8 * half;
            BV ah0, al0, ah1, al1;
            ah0.q[0] = *reinterpret_cast<const uint4*>(&Thi[ofs0]);
            ah0.q[1] = *reinterpret_cast<const uint4*>(&Thi[ofs0 + 4]);
            al0.q[0] = *reinterpret_cast<const uint4*>(&Tlo[ofs0]);
            al0.q[1] = *reinterpret_cast<const uint4*>(&Tlo[ofs0 + 4]);
            ah1.q[0] = *reinterpret_cast<const uint4*>(&Thi[ofs1]);
            ah1.q[1] = *reinterpret_cast<const uint4*>(&Thi[ofs1 + 4]);
            al1.q[0] = *reinterpret_cast<const uint4*>(&Tlo[ofs1]);
            al1.q[1] = *reinterpret_cast<const uint4*>(&Tlo[ofs1 + 4]);
            // B = H tile: sign(e) = w16(e) ^ parity(kc&wave) ^ (q&half)
            const unsigned P2 = (unsigned)(__builtin_popcount(kc & wave) & 1);
            const unsigned SB = (w16 ^ (0u - P2)) & 0xFFFFu;
            const unsigned qm = half ? 0x80008000u : 0u;
            BV b0, b1;
            #pragma unroll
            for (int p = 0; p < 8; ++p) {
                unsigned e0 = (SB >> (2 * p)) & 1u, e1 = (SB >> (2 * p + 1)) & 1u;
                b0.u[p] = HBF | (e0 << 15) | (e1 << 31);
                b1.u[p] = b0.u[p] ^ qm;
            }
            WMMA(o[0][0], ah0, b0); WMMA(o[0][0], al0, b0);
            WMMA(o[0][1], ah0, b1); WMMA(o[0][1], al0, b1);
            WMMA(o[1][0], ah1, b0); WMMA(o[1][0], al1, b0);
            WMMA(o[1][1], ah1, b1); WMMA(o[1][1], al1, b1);
        }
        // fused affine epilogue + coalesced store
        #pragma unroll
        for (int mt = 0; mt < 2; ++mt)
            #pragma unroll
            for (int q = 0; q < 2; ++q) {
                const int col = (wave * 2 + q) * 16 + l15;
                #pragma unroll
                for (int r = 0; r < 8; ++r) {
                    const int row = strip * 32 + mt * 16 + r + 8 * half;
                    float val = o[mt][q][r];
                    if (AFFINE) val = s * val + t;
                    dst[base + (size_t)row * IMGW + col] = val;
                }
            }
        __syncthreads();   // T buffers reused next strip
    }
}

extern "C" void kernel_launch(void* const* d_in, const int* in_sizes, int n_in,
                              void* d_out, int out_size, void* d_ws, size_t ws_size,
                              hipStream_t stream) {
    const float* x  = (const float*)d_in[0];
    const float* cw = (const float*)d_in[1];
    const float* cb = (const float*)d_in[2];
    const float* ow = (const float*)d_in[3];
    const float* ob = (const float*)d_in[4];
    float* out = (float*)d_out;

    // 256 patches -> 256 workgroups of 256 threads (8 waves, 1 WG/WGP at this LDS).
    hadamard_pass<true ><<<256, 256, 0, stream>>>(x,   out, cw, cb, ow, ob);
    // Pass B reads the 64MB intermediate straight from the 192MB L2.
    hadamard_pass<false><<<256, 256, 0, stream>>>(out, out, cw, cb, ow, ob);
}
